// SimpleS4_20993800142892
// MI455X (gfx1250) — matrix operand order repaired and checked by reading, not compiled, not run on hardware
//
#include <hip/hip_runtime.h>

typedef __attribute__((ext_vector_type(16))) __bf16        v16bf;
typedef __attribute__((ext_vector_type(8)))  float         v8f;
typedef __attribute__((ext_vector_type(8)))  unsigned int  v8u;

#define NB  8
#define NH  256
#define NL  4096
#define NDD 32
#define QC  64
#define NCHUNK (NL / QC)
#define GUARD 64

// split f32 into bf16 hi + bf16 lo(residual), packed {lo[31:16], hi[15:0]}
__device__ __forceinline__ unsigned pack_split(float x) {
    __bf16 hb = (__bf16)x;
    float  r  = x - (float)hb;
    __bf16 lb = (__bf16)r;
    unsigned h = (unsigned)__builtin_bit_cast(unsigned short, hb);
    unsigned l = (unsigned)__builtin_bit_cast(unsigned short, lb);
    return h | (l << 16);
}

// ---------------------------------------------------------------------------
// Kernel 1: build filter f[h][l] (packed split-bf16) and init[h][l] (f32).
// ---------------------------------------------------------------------------
__global__ __launch_bounds__(256)
void s4_build_filter(const float* __restrict__ theta,
                     const float* __restrict__ a,
                     const float* __restrict__ b_p,
                     const float* __restrict__ c_p,
                     const float* __restrict__ x0,
                     unsigned* __restrict__ fpack,
                     float* __restrict__ ginit)
{
    const int h = blockIdx.y;
    const int l = blockIdx.x * blockDim.x + threadIdx.x;
    const float T = 1.0f / (float)(NL - 1);
    const float z = T * (float)l;
    float facc = 0.0f, gacc = 0.0f;
    #pragma unroll 4
    for (int d = 0; d < NDD; ++d) {
        const int   idx = h * NDD + d;
        const float th  = theta[idx];
        const float aa  = fabsf(a[idx]);
        const float bp  = b_p[idx];
        const float cp  = c_p[idx];
        const float xv  = x0[idx];
        // term0 = (exp(ls*T) - 1) / ls,  ls = -aa + i*th
        const float eT = __expf(-aa * T);
        const float cT = __cosf(th * T);
        const float sT = __sinf(th * T);
        const float nr = eT * cT - 1.0f;
        const float ni = eT * sT;
        const float den = aa * aa + th * th;   // > 0 (aa = h_scale > 0)
        const float t0r = (-aa * nr + th * ni) / den;
        const float t0i = (-aa * ni - th * nr) / den;
        const float ez  = __expf(-aa * z);
        const float cz  = __cosf(th * z);
        const float sz  = __sinf(th * z);
        facc += (bp * cp) * 2.0f * ez * (t0r * cz - t0i * sz);
        gacc += 2.0f * (cp * xv) * 2.0f * T * ez * cz;
    }
    fpack[h * NL + l] = pack_split(facc);
    ginit[h * NL + l] = gacc;
}

// ---------------------------------------------------------------------------
// Kernel 2: pre-split u into packed {bf16_lo, bf16_hi} words (same layout).
// ---------------------------------------------------------------------------
__global__ __launch_bounds__(256)
void s4_pack_u(const float* __restrict__ u, unsigned* __restrict__ upack)
{
    const size_t i = (size_t)blockIdx.x * 256 + threadIdx.x;
    upack[i] = pack_split(u[i]);
}

// ---------------------------------------------------------------------------
// Kernel 3: blocked Toeplitz causal convolution, v_wmma_f32_16x16x32_bf16
// with split-bf16 operands (3 WMMAs per K-slab -> ~f32 accuracy).
// WG = (channel h, output chunk ic); 2 waves x 2 M-subtiles each, so one
// B-fragment build feeds 6 WMMAs. f[h][:] packed in LDS with a 64-word
// zero guard => unconditional band gathers. Padded B columns 8..15 are dead.
// ---------------------------------------------------------------------------
__global__ __launch_bounds__(64)
void s4_conv(const unsigned* __restrict__ upack,
             const float* __restrict__ u,
             const unsigned* __restrict__ fpack,
             const float* __restrict__ ginit,
             const float* __restrict__ Dvec,
             float* __restrict__ out)
{
    __shared__ unsigned fsh[GUARD + NL];       // 16.25 KB
    const int h    = blockIdx.y;
    const int ic   = blockIdx.x;
    const int tid  = threadIdx.x;
    const int wid  = tid >> 5;                 // wave 0/1 -> subtiles {0,1}/{2,3}
    const int lane = tid & 31;

    // zero guard + cooperative filter load
    if (tid < GUARD) fsh[tid] = 0u;
    {
        const uint4* fsrc = (const uint4*)(fpack + (size_t)h * NL);
        uint4* fdst = (uint4*)(fsh + GUARD);
        for (int t = tid; t < NL / 4; t += 64) fdst[t] = fsrc[t];
    }
    __syncthreads();

    const int n   = lane & 15;                 // N column = batch (8..15 dead)
    const int hi2 = lane >> 4;
    const int m   = lane & 15;                 // A-fragment row within subtile
    const int nc  = (n < NB) ? n : (NB - 1);   // keep addresses valid
    const unsigned* ucolp = upack + ((size_t)nc * NH + h) * NL;

    v8f acc0 = {0.f, 0.f, 0.f, 0.f, 0.f, 0.f, 0.f, 0.f};
    v8f acc1 = {0.f, 0.f, 0.f, 0.f, 0.f, 0.f, 0.f, 0.f};

    for (int j = 0; j <= ic; ++j) {
        // A row index base for subtile pair (wid*2, wid*2+1)
        const int fbase0 = (ic - j) * QC + (wid * 2) * 16 + m + GUARD;
        #pragma unroll
        for (int kh = 0; kh < 2; ++kh) {
            // ---- shared B fragment: 16 consecutive packed u words, column n
            const int l0 = j * QC + kh * 32 + hi2 * 16;
            unsigned uw[16];
            const uint4* up = (const uint4*)(ucolp + l0);
            #pragma unroll
            for (int e4 = 0; e4 < 4; ++e4) {
                uint4 v = up[e4];
                uw[e4 * 4 + 0] = v.x; uw[e4 * 4 + 1] = v.y;
                uw[e4 * 4 + 2] = v.z; uw[e4 * 4 + 3] = v.w;
            }
            v8u bhw, blw;
            #pragma unroll
            for (int w = 0; w < 8; ++w) {
                bhw[w] = __builtin_amdgcn_perm(uw[2*w+1], uw[2*w], 0x05040100u);
                blw[w] = __builtin_amdgcn_perm(uw[2*w+1], uw[2*w], 0x07060302u);
            }
            const v16bf bhi = __builtin_bit_cast(v16bf, bhw);
            const v16bf blo = __builtin_bit_cast(v16bf, blw);

            // ---- two A fragments (subtile s=0,1): band gather from LDS
            unsigned f0[16], f1[16];
            #pragma unroll
            for (int e = 0; e < 16; ++e) {
                const int k = e + hi2 * 16 + kh * 32;
                f0[e] = fsh[fbase0      - k];
                f1[e] = fsh[fbase0 + 16 - k];
            }
            v8u a0h, a0l, a1h, a1l;
            #pragma unroll
            for (int w = 0; w < 8; ++w) {
                a0h[w] = __builtin_amdgcn_perm(f0[2*w+1], f0[2*w], 0x05040100u);
                a0l[w] = __builtin_amdgcn_perm(f0[2*w+1], f0[2*w], 0x07060302u);
                a1h[w] = __builtin_amdgcn_perm(f1[2*w+1], f1[2*w], 0x05040100u);
                a1l[w] = __builtin_amdgcn_perm(f1[2*w+1], f1[2*w], 0x07060302u);
            }
            const v16bf ahi0 = __builtin_bit_cast(v16bf, a0h);
            const v16bf alo0 = __builtin_bit_cast(v16bf, a0l);
            const v16bf ahi1 = __builtin_bit_cast(v16bf, a1h);
            const v16bf alo1 = __builtin_bit_cast(v16bf, a1l);

            // split product: u_hi*f_hi + u_lo*f_hi + u_hi*f_lo (f32 accum);
            // interleave the two independent accumulator chains
            acc0 = __builtin_amdgcn_wmma_f32_16x16x32_bf16(false, ahi0, false, bhi,
                                                           (short)0, acc0, false, false);
            acc1 = __builtin_amdgcn_wmma_f32_16x16x32_bf16(false, ahi1, false, bhi,
                                                           (short)0, acc1, false, false);
            acc0 = __builtin_amdgcn_wmma_f32_16x16x32_bf16(false, ahi0, false, blo,
                                                           (short)0, acc0, false, false);
            acc1 = __builtin_amdgcn_wmma_f32_16x16x32_bf16(false, ahi1, false, blo,
                                                           (short)0, acc1, false, false);
            acc0 = __builtin_amdgcn_wmma_f32_16x16x32_bf16(false, alo0, false, bhi,
                                                           (short)0, acc0, false, false);
            acc1 = __builtin_amdgcn_wmma_f32_16x16x32_bf16(false, alo1, false, bhi,
                                                           (short)0, acc1, false, false);
        }
    }

    // epilogue: y += u*D[h] + init[h][l]
    const float Dh = Dvec[h];
    if (n < NB) {
        #pragma unroll
        for (int s = 0; s < 2; ++s) {
            const v8f acc = s ? acc1 : acc0;
            #pragma unroll
            for (int r = 0; r < 8; ++r) {
                const int p = (wid * 2 + s) * 16 + r + 8 * hi2;  // M = r + 8*hi
                const int l = ic * QC + p;
                const size_t oidx = ((size_t)n * NH + h) * NL + l;
                out[oidx] = acc[r] + u[oidx] * Dh + ginit[(size_t)h * NL + l];
            }
        }
    }
}

// ---------------------------------------------------------------------------
extern "C" void kernel_launch(void* const* d_in, const int* in_sizes, int n_in,
                              void* d_out, int out_size, void* d_ws, size_t ws_size,
                              hipStream_t stream) {
    (void)in_sizes; (void)n_in; (void)out_size; (void)ws_size;
    const float* u     = (const float*)d_in[0];
    const float* theta = (const float*)d_in[1];
    const float* a     = (const float*)d_in[2];
    const float* Dv    = (const float*)d_in[3];
    const float* b_p   = (const float*)d_in[4];
    const float* c_p   = (const float*)d_in[5];
    const float* x0    = (const float*)d_in[6];
    float* out = (float*)d_out;

    unsigned* fpack = (unsigned*)d_ws;                         // H*L u32 (4 MB)
    float*    ginit = (float*)(fpack + (size_t)NH * NL);       // H*L f32 (4 MB)
    unsigned* upack = (unsigned*)(ginit + (size_t)NH * NL);    // B*H*L u32 (33.5 MB)

    dim3 g1(NL / 256, NH), b1(256);
    s4_build_filter<<<g1, b1, 0, stream>>>(theta, a, b_p, c_p, x0, fpack, ginit);

    const int nu = NB * NH * NL;
    s4_pack_u<<<nu / 256, 256, 0, stream>>>(u, upack);

    dim3 g2(NCHUNK, NH), b2(64);
    s4_conv<<<g2, b2, 0, stream>>>(upack, u, fpack, ginit, Dv, out);
}
